// NodeShuffle_60610578481739
// MI455X (gfx1250) — matrix operand order repaired and checked by reading, not compiled, not run on hardware
//
#include <hip/hip_runtime.h>

#define BB    2
#define NPTS  4096
#define CIN   32
#define EMBD  1024
#define KNBR  16
#define UPF   16
#define BNEPS 1e-5f
#define ROWS_PER_BLK 128

typedef __attribute__((ext_vector_type(16))) _Float16 v16h;
typedef __attribute__((ext_vector_type(8)))  float    v8f;

// ---------------------------------------------------------------------------
// Async global->LDS helpers (gfx1250 GLOBAL_LOAD_ASYNC_TO_LDS_*, ASYNCcnt).
// The instruction's VDST VGPR holds the workgroup-relative LDS byte offset:
//   the LDS address is the LDS base plus this offset plus the instruction offset
// We recover that offset via an addrspace(3) cast of the __shared__ pointer.
// ---------------------------------------------------------------------------
typedef __attribute__((address_space(3))) _Float16 lds_h;

__device__ __forceinline__ unsigned lds_off(const _Float16* p) {
  return (unsigned)(unsigned long long)(lds_h*)p;
}
__device__ __forceinline__ void async_copy_b64(unsigned ldsDst, const _Float16* gsrc) {
  asm volatile("global_load_async_to_lds_b64 %0, %1, off"
               :: "v"(ldsDst), "v"(gsrc) : "memory");
}
__device__ __forceinline__ void async_copy_b32(unsigned ldsDst, const _Float16* gsrc) {
  asm volatile("global_load_async_to_lds_b32 %0, %1, off"
               :: "v"(ldsDst), "v"(gsrc) : "memory");
}
__device__ __forceinline__ void wait_async0() {
  asm volatile("s_wait_asynccnt 0x0" ::: "memory");
}

// ---------------------------------------------------------------------------
// KNN: brute force with LDS candidate tiles, register-resident sorted top-16.
// ---------------------------------------------------------------------------
__global__ void __launch_bounds__(256)
knn_kernel(const float* __restrict__ xyz, int* __restrict__ knn_idx) {
  const int b = blockIdx.y;
  const int n = blockIdx.x * 256 + threadIdx.x;
  const float* base = xyz + (size_t)b * NPTS * 3;
  const float px = base[n * 3 + 0];
  const float py = base[n * 3 + 1];
  const float pz = base[n * 3 + 2];

  float bd[KNBR]; int bi[KNBR];
#pragma unroll
  for (int t = 0; t < KNBR; ++t) { bd[t] = 3.0e38f; bi[t] = 0; }

  __shared__ float sx[256], sy[256], sz[256];
  for (int t0 = 0; t0 < NPTS; t0 += 256) {
    __syncthreads();
    const int c = t0 + threadIdx.x;
    sx[threadIdx.x] = base[c * 3 + 0];
    sy[threadIdx.x] = base[c * 3 + 1];
    sz[threadIdx.x] = base[c * 3 + 2];
    __syncthreads();
    for (int j = 0; j < 256; ++j) {
      const float dx = px - sx[j], dy = py - sy[j], dz = pz - sz[j];
      const float d = dx * dx + dy * dy + dz * dz;
      if (d < bd[KNBR - 1]) {         // insertion into sorted list, static idx
        float cd = d; int ci = t0 + j;
#pragma unroll
        for (int t = 0; t < KNBR; ++t) {
          if (cd < bd[t]) {
            float td = bd[t]; int ti = bi[t];
            bd[t] = cd; bi[t] = ci; cd = td; ci = ti;
          }
        }
      }
    }
  }
  int* o = knn_idx + (size_t)(b * NPTS + n) * KNBR;
#pragma unroll
  for (int t = 0; t < KNBR; ++t) o[t] = bi[t];
}

// ---------------------------------------------------------------------------
// Packing / weight-prep kernels
// ---------------------------------------------------------------------------
__global__ void pack_feature_kernel(const float* __restrict__ f,
                                    _Float16* __restrict__ xh) {
  const int i = blockIdx.x * 256 + threadIdx.x;      // over B*N*CIN
  if (i >= BB * NPTS * CIN) return;
  const int c = i % CIN;
  const int n = (i / CIN) % NPTS;
  const int b = i / (CIN * NPTS);
  xh[i] = (_Float16)f[((size_t)b * CIN + c) * NPTS + n];
}

// W:(EMBD, 2C) -> Wl' = W[:, :C]-W[:, C:]  and  Wr = W[:, C:], both f16 (EMBD,C)
__global__ void prep_w_kernel(const float* __restrict__ W,
                              _Float16* __restrict__ Wl,
                              _Float16* __restrict__ Wr, int C) {
  const int i = blockIdx.x * 256 + threadIdx.x;      // over EMBD*C
  if (i >= EMBD * C) return;
  const int o = i / C, c = i % C;
  const float wr = W[(size_t)o * 2 * C + C + c];
  const float wl = W[(size_t)o * 2 * C + c] - wr;
  Wl[i] = (_Float16)wl;
  Wr[i] = (_Float16)wr;
}

__global__ void pack_h_kernel(const float* __restrict__ src,
                              _Float16* __restrict__ dst, int n) {
  const int i = blockIdx.x * 256 + threadIdx.x;
  if (i < n) dst[i] = (_Float16)src[i];
}

__global__ void fill0_kernel(float* __restrict__ p, int n) {
  const int i = blockIdx.x * 256 + threadIdx.x;
  if (i < n) p[i] = 0.0f;
}

// ---------------------------------------------------------------------------
// Fused per-point GEMM:  u = A * Wl^T (+bias),  v = A * Wr^T
// A:(M,Kdim) f16 row-major, Wl/Wr:(EMBD,Kdim) f16 row-major.
// Block = 8 waves; block tile = 128 M-rows x 32 out-channels x {u,v}.
// One A fragment feeds 4 WMMAs (2 o-subtiles x 2 weight matrices).
// W tiles staged with async global->LDS copies (ASYNCcnt path).
// A fragment lane layout (16-bit A 16x32): lane L -> row L%16, half h=L/16,
// VGPR v holds pair at k = 16h + 2v.
// ---------------------------------------------------------------------------
__global__ void __launch_bounds__(256)
gemm_uv_kernel(const _Float16* __restrict__ A, int lda, int Kdim,
               const _Float16* __restrict__ Wl, const _Float16* __restrict__ Wr,
               const float* __restrict__ bias,
               _Float16* __restrict__ uout, _Float16* __restrict__ vout) {
  __shared__ __align__(16) _Float16 sBl[32 * 32];    // [o][k]
  __shared__ __align__(16) _Float16 sBr[32 * 32];
  const int lane = threadIdx.x & 31;
  const int wave = threadIdx.x >> 5;
  const int m0 = blockIdx.x * 128 + wave * 16;
  const int o0 = blockIdx.y * 32;
  const int row = m0 + (lane & 15);
  const int h = lane >> 4;

  // per-thread staging slot (constant across K loop)
  const int so  = threadIdx.x & 31;
  const int sk4 = (threadIdx.x >> 5) << 2;           // 4 halfs per thread
  const unsigned dstL = lds_off(&sBl[so * 32 + sk4]);
  const unsigned dstR = lds_off(&sBr[so * 32 + sk4]);

  v8f accU0 = {0.f,0.f,0.f,0.f,0.f,0.f,0.f,0.f};
  v8f accU1 = accU0, accV0 = accU0, accV1 = accU0;

  for (int k0 = 0; k0 < Kdim; k0 += 32) {
    { // async cooperative stage of both 32x32 W tiles (global -> LDS, no VGPR trip)
      const size_t woff = (size_t)(o0 + so) * Kdim + k0 + sk4;
      async_copy_b64(dstL, Wl + woff);
      async_copy_b64(dstR, Wr + woff);
      wait_async0();
    }
    __syncthreads();

    union { v16h h16; unsigned u[8]; } af;
    const unsigned* arow = (const unsigned*)(A + (size_t)row * lda + k0);
#pragma unroll
    for (int v = 0; v < 8; ++v) {
      const int kb = 16 * h + 2 * v;
      af.u[v] = arow[kb >> 1];
    }
    __builtin_prefetch(arow + 16, 0, 1);             // next K-chunk of this row

    auto ldb = [&](const _Float16* sB, int s) -> v16h {
      union { v16h h16; unsigned u[8]; } bf;
      const unsigned* bsrc =
          (const unsigned*)(sB + ((lane & 15) + s) * 32 + 16 * h);
#pragma unroll
      for (int v = 0; v < 8; ++v) bf.u[v] = bsrc[v];
      return bf.h16;
    };
    accU0 = __builtin_amdgcn_wmma_f32_16x16x32_f16(false, af.h16, false, ldb(sBl, 0),
                                                   (short)0, accU0, false, false);
    accU1 = __builtin_amdgcn_wmma_f32_16x16x32_f16(false, af.h16, false, ldb(sBl, 16),
                                                   (short)0, accU1, false, false);
    accV0 = __builtin_amdgcn_wmma_f32_16x16x32_f16(false, af.h16, false, ldb(sBr, 0),
                                                   (short)0, accV0, false, false);
    accV1 = __builtin_amdgcn_wmma_f32_16x16x32_f16(false, af.h16, false, ldb(sBr, 16),
                                                   (short)0, accV1, false, false);
    __syncthreads();
  }

  // C/D layout: VGPR r, lane L -> m = r + 8*(L/16), n = L%16.
  const int n0 = o0 + (lane & 15);
  const int n1 = n0 + 16;
  const float bias0 = bias[n0], bias1 = bias[n1];
#pragma unroll
  for (int r = 0; r < 8; ++r) {
    const int m = m0 + r + 8 * h;
    uout[(size_t)m * EMBD + n0] = (_Float16)(accU0[r] + bias0);  // bias folded into u
    uout[(size_t)m * EMBD + n1] = (_Float16)(accU1[r] + bias1);
    vout[(size_t)m * EMBD + n0] = (_Float16)accV0[r];
    vout[(size_t)m * EMBD + n1] = (_Float16)accV1[r];
  }
}

// ---------------------------------------------------------------------------
// Edge reduction pass A: per-channel sum / sumsq of y = u[i] + v[j]
// (conv bias already folded into u by the GEMM epilogue)
// ---------------------------------------------------------------------------
__global__ void __launch_bounds__(256)
edge_stats_kernel(const _Float16* __restrict__ u, const _Float16* __restrict__ v,
                  const int* __restrict__ knn_idx,
                  float* __restrict__ sum, float* __restrict__ sumsq) {
  __shared__ int sIdx[KNBR];
  const int ch = blockIdx.x * 256 + threadIdx.x;
  const int r0 = blockIdx.y * ROWS_PER_BLK;
  float s = 0.f, s2 = 0.f;
  for (int r = r0; r < r0 + ROWS_PER_BLK; ++r) {
    __syncthreads();
    if (threadIdx.x < KNBR) sIdx[threadIdx.x] = knn_idx[(size_t)r * KNBR + threadIdx.x];
    __syncthreads();
    const int bb = (r / NPTS) * NPTS;
    const float ub = (float)u[(size_t)r * EMBD + ch];
#pragma unroll 4
    for (int k = 0; k < KNBR; ++k) {
      const float y = ub + (float)v[(size_t)(bb + sIdx[k]) * EMBD + ch];
      s += y; s2 += y * y;
    }
  }
  atomicAdd(&sum[ch], s);
  atomicAdd(&sumsq[ch], s2);
}

__global__ void finalize_stats_kernel(const float* __restrict__ sum,
                                      const float* __restrict__ sumsq,
                                      const float* __restrict__ g,
                                      const float* __restrict__ be,
                                      float* __restrict__ scale,
                                      float* __restrict__ shift) {
  const int i = blockIdx.x * 256 + threadIdx.x;
  if (i >= EMBD) return;
  const float inv = 1.0f / (float)(BB * NPTS * KNBR);
  const float mu  = sum[i] * inv;
  const float var = sumsq[i] * inv - mu * mu;
  const float sc  = g[i] * rsqrtf(var + BNEPS);
  scale[i] = sc;
  shift[i] = be[i] - mu * sc;
}

// ---------------------------------------------------------------------------
// Edge reduction pass B: out[i,ch] = max_k relu((u+v[j])*scale + shift)
// ---------------------------------------------------------------------------
__global__ void __launch_bounds__(256)
edge_max_kernel(const _Float16* __restrict__ u, const _Float16* __restrict__ v,
                const int* __restrict__ knn_idx,
                const float* __restrict__ scale, const float* __restrict__ shift,
                _Float16* __restrict__ xout) {
  __shared__ int sIdx[KNBR];
  const int ch = blockIdx.x * 256 + threadIdx.x;
  const int r0 = blockIdx.y * ROWS_PER_BLK;
  const float sc = scale[ch], sh = shift[ch];
  for (int r = r0; r < r0 + ROWS_PER_BLK; ++r) {
    __syncthreads();
    if (threadIdx.x < KNBR) sIdx[threadIdx.x] = knn_idx[(size_t)r * KNBR + threadIdx.x];
    __syncthreads();
    const int bb = (r / NPTS) * NPTS;
    const float ub = (float)u[(size_t)r * EMBD + ch];
    float m = 0.0f;                                  // relu >= 0
#pragma unroll 4
    for (int k = 0; k < KNBR; ++k) {
      const float y = ub + (float)v[(size_t)(bb + sIdx[k]) * EMBD + ch];
      m = fmaxf(m, fmaf(y, sc, sh));
    }
    xout[(size_t)r * EMBD + ch] = (_Float16)m;
  }
}

// ---------------------------------------------------------------------------
// Final projection: out(8192,48) = X(8192,1024) * Wp(48,1024)^T, fused
// bias + xyz epilogue writing (B, N*UP, 3).
// ---------------------------------------------------------------------------
__global__ void __launch_bounds__(256)
final_gemm_kernel(const _Float16* __restrict__ A,
                  const _Float16* __restrict__ Wp,
                  const float* __restrict__ bp,
                  const float* __restrict__ xyz,
                  float* __restrict__ out) {
  __shared__ __align__(16) _Float16 sB[16 * 32];
  const int lane = threadIdx.x & 31;
  const int wave = threadIdx.x >> 5;
  const int m0 = (blockIdx.x * 8 + wave) * 16;
  const int o0 = blockIdx.y * 16;                    // 3 tiles cover 48 channels
  const int row = m0 + (lane & 15);
  const int h = lane >> 4;

  const int sn  = threadIdx.x & 15;
  const int sk2 = (threadIdx.x >> 4) << 1;
  const unsigned dstB = lds_off(&sB[sn * 32 + sk2]);

  v8f acc = {0.f, 0.f, 0.f, 0.f, 0.f, 0.f, 0.f, 0.f};

  for (int k0 = 0; k0 < EMBD; k0 += 32) {
    async_copy_b32(dstB, Wp + (size_t)(o0 + sn) * EMBD + k0 + sk2);
    wait_async0();
    __syncthreads();
    union { v16h h16; unsigned u[8]; } af, bf;
    const unsigned* arow = (const unsigned*)(A + (size_t)row * EMBD + k0);
#pragma unroll
    for (int v = 0; v < 8; ++v) {
      const int kb = 16 * h + 2 * v;
      af.u[v] = arow[kb >> 1];
    }
    __builtin_prefetch(arow + 16, 0, 1);
    {
      const unsigned* bsrc = (const unsigned*)(&sB[(lane & 15) * 32 + 16 * h]);
#pragma unroll
      for (int v = 0; v < 8; ++v) bf.u[v] = bsrc[v];
    }
    acc = __builtin_amdgcn_wmma_f32_16x16x32_f16(false, af.h16, false, bf.h16,
                                                 (short)0, acc, false, false);
    __syncthreads();
  }

  const int ch = o0 + (lane & 15);                   // 0..47
  const int up = ch / 3, d = ch - up * 3;
#pragma unroll
  for (int r = 0; r < 8; ++r) {
    const int m = m0 + r + 8 * h;
    const int b = m / NPTS, n = m % NPTS;
    const float val = acc[r] + bp[ch] + xyz[(size_t)(b * NPTS + n) * 3 + d];
    out[((size_t)(b * NPTS + n) * UPF + up) * 3 + d] = val;
  }
}

// ---------------------------------------------------------------------------
// Launcher
// ---------------------------------------------------------------------------
extern "C" void kernel_launch(void* const* d_in, const int* in_sizes, int n_in,
                              void* d_out, int out_size, void* d_ws, size_t ws_size,
                              hipStream_t stream) {
  const float* xyz  = (const float*)d_in[0];
  const float* feat = (const float*)d_in[1];
  const float* W1   = (const float*)d_in[2];
  const float* b1   = (const float*)d_in[3];
  const float* g1   = (const float*)d_in[4];
  const float* be1  = (const float*)d_in[5];
  const float* W2   = (const float*)d_in[6];
  const float* b2   = (const float*)d_in[7];
  const float* g2   = (const float*)d_in[8];
  const float* be2  = (const float*)d_in[9];
  const float* Wp   = (const float*)d_in[10];
  const float* bp   = (const float*)d_in[11];
  float* out = (float*)d_out;

  // ---- workspace carve (~53 MB) ----
  char* ws = (char*)d_ws;
  size_t off = 0;
  auto carve = [&](size_t bytes) -> void* {
    void* p = ws + off;
    off = (off + bytes + 255) & ~(size_t)255;
    return p;
  };
  const size_t M = (size_t)BB * NPTS;                      // 8192 point rows
  int*       idx   = (int*)      carve(M * KNBR * sizeof(int));
  _Float16*  xh    = (_Float16*) carve(M * EMBD * sizeof(_Float16));  // x (f16), reused per layer
  _Float16*  ubuf  = (_Float16*) carve(M * EMBD * sizeof(_Float16));
  _Float16*  vbuf  = (_Float16*) carve(M * EMBD * sizeof(_Float16));
  _Float16*  Wl    = (_Float16*) carve((size_t)EMBD * EMBD * sizeof(_Float16));
  _Float16*  Wr    = (_Float16*) carve((size_t)EMBD * EMBD * sizeof(_Float16));
  _Float16*  wph   = (_Float16*) carve((size_t)3 * UPF * EMBD * sizeof(_Float16));
  float*     sum   = (float*)    carve(2 * EMBD * sizeof(float));     // sum | sumsq
  float*     sumsq = sum + EMBD;
  float*     scale = (float*)    carve(2 * EMBD * sizeof(float));     // scale | shift
  float*     shift = scale + EMBD;
  (void)ws_size; (void)in_sizes; (void)n_in; (void)out_size;

  // ---- KNN + input packing ----
  knn_kernel<<<dim3(NPTS / 256, BB), 256, 0, stream>>>(xyz, idx);
  pack_feature_kernel<<<(BB * NPTS * CIN) / 256, 256, 0, stream>>>(feat, xh);

  // ---- EdgeConv layer 1 (C = 32) ----
  prep_w_kernel<<<(EMBD * CIN + 255) / 256, 256, 0, stream>>>(W1, Wl, Wr, CIN);
  gemm_uv_kernel<<<dim3(M / 128, EMBD / 32), 256, 0, stream>>>(
      xh, CIN, CIN, Wl, Wr, b1, ubuf, vbuf);
  fill0_kernel<<<(2 * EMBD + 255) / 256, 256, 0, stream>>>(sum, 2 * EMBD);
  edge_stats_kernel<<<dim3(EMBD / 256, M / ROWS_PER_BLK), 256, 0, stream>>>(
      ubuf, vbuf, idx, sum, sumsq);
  finalize_stats_kernel<<<(EMBD + 255) / 256, 256, 0, stream>>>(sum, sumsq, g1, be1, scale, shift);
  edge_max_kernel<<<dim3(EMBD / 256, M / ROWS_PER_BLK), 256, 0, stream>>>(
      ubuf, vbuf, idx, scale, shift, xh);                   // xh = layer-1 output (f16)

  // ---- EdgeConv layer 2 (C = 1024) ----
  prep_w_kernel<<<(EMBD * EMBD + 255) / 256, 256, 0, stream>>>(W2, Wl, Wr, EMBD);
  gemm_uv_kernel<<<dim3(M / 128, EMBD / 32), 256, 0, stream>>>(
      xh, EMBD, EMBD, Wl, Wr, b2, ubuf, vbuf);
  fill0_kernel<<<(2 * EMBD + 255) / 256, 256, 0, stream>>>(sum, 2 * EMBD);
  edge_stats_kernel<<<dim3(EMBD / 256, M / ROWS_PER_BLK), 256, 0, stream>>>(
      ubuf, vbuf, idx, sum, sumsq);
  finalize_stats_kernel<<<(EMBD + 255) / 256, 256, 0, stream>>>(sum, sumsq, g2, be2, scale, shift);
  edge_max_kernel<<<dim3(EMBD / 256, M / ROWS_PER_BLK), 256, 0, stream>>>(
      ubuf, vbuf, idx, scale, shift, xh);                   // xh = layer-2 output (f16)

  // ---- Final projection + fused epilogue ----
  pack_h_kernel<<<(3 * UPF * EMBD + 255) / 256, 256, 0, stream>>>(Wp, wph, 3 * UPF * EMBD);
  final_gemm_kernel<<<dim3(M / 128, 3), 256, 0, stream>>>(xh, wph, bp, xyz, out);
}
